// CIN_81544249082266
// MI455X (gfx1250) — matrix operand order repaired and checked by reading, not compiled, not run on hardware
//
#include <hip/hip_runtime.h>

// ---------------------------------------------------------------------------
// CIN (xDeepFM) forward for MI455X (gfx1250, wave32, WMMA, bf16/f32).
// Unified K-indexing c = i*64 + j for ALL layers (layer-0 j padded 39->64,
// triu mask * 2 folded into pre-swizzled filter fragments).
// Per block: 8 waves = 8 samples; B fragments staged once per K-step into LDS
// and shared by all waves (8x L2-traffic reduction), with a register prefetch
// pipeline (peeled last step, pointer-increment addressing, no unroll) so L2
// latency hides behind WMMAs without blowing past 256 VGPRs.
// A fragments built from d-major transposed bf16 x/hidden via v_pk_mul_bf16.
// ---------------------------------------------------------------------------

typedef __attribute__((ext_vector_type(2)))  __bf16 v2bf;
typedef __attribute__((ext_vector_type(16))) __bf16 v16bf;
typedef __attribute__((ext_vector_type(8)))  float  v8f;

#define NF0   39
#define DD    16
#define NB    8192
#define XLEN  (NF0*DD)          // 624 floats per sample

#define KT       78             // 2496 / 32 K-steps, all layers
#define FRAG_HALVES_PER_KS 4096 // 8 ntiles * 32 lanes * 16 halves
#define FRAG_UINT4_PER_KS  512  // 8KB per K-step
#define HSTRIDE  72             // halves per d-row (bank-spread, 16B aligned)
#define SMP_H    (DD*HSTRIDE)   // 1152 halves per buffer

__device__ __forceinline__ unsigned short f2bf(float f) {
  unsigned x = __float_as_uint(f);
  unsigned r = x + 0x7FFFu + ((x >> 16) & 1u);   // round-to-nearest-even
  return (unsigned short)(r >> 16);
}

// ---------------------------------------------------------------------------
// Prep: filter (c_len x 128 fp32) -> bf16 fragments, WMMA-B lane layout,
// unified K index c = i*64 + j:
//   halves[((ks*8+nt)*32+lane)*16+p],  k = (p<8?p:p+8) + 8*(lane>=16),  c = ks*32+k,
//   l = nt*16 + (lane&15),  i = c>>6, j = c&63.
// mode 0 (layer 0): src row = i*39+j, value = (j<39 && i<j) ? 2*f : 0.
// mode 1 (layers 1/2): src row = c directly (field 64 native).
// ---------------------------------------------------------------------------
__global__ void prep_filter(const float* __restrict__ src,
                            unsigned short* __restrict__ dst, int mode) {
  int t = blockIdx.x * 256 + threadIdx.x;
  if (t >= KT * FRAG_HALVES_PER_KS) return;
  int p    = t & 15;
  int lane = (t >> 4) & 31;
  int nt   = (t >> 9) & 7;
  int ks   = t >> 12;
  int k    = (p < 8 ? p : p + 8) + ((lane & 16) ? 8 : 0);
  int c    = ks * 32 + k;
  int i    = c >> 6;
  int j    = c & 63;
  int l    = nt * 16 + (lane & 15);
  float v = 0.0f;
  if (mode == 0) {
    if (j < NF0 && i < j) v = src[(size_t)(i * NF0 + j) * 128 + l] * 2.0f;
  } else {
    v = src[(size_t)c * 128 + l];
  }
  dst[t] = f2bf(v);
}

// ---------------------------------------------------------------------------
// One CIN layer for one wave (one sample), B staged per-block in LDS with a
// one-K-step register prefetch pipeline (last step peeled).
// A layout (16-bit 16x32): lane L: M = L&15; half p -> K=(p<8?p:p+8)+8*(L>=16).
// Groups of 8 halves share i (x scalar) and read 8 consecutive j (one b128).
// ---------------------------------------------------------------------------
template <int KEEPHALF>
__device__ void run_layer(const unsigned short* __restrict__ xT,
                          const unsigned short* __restrict__ hT,
                          const unsigned short* __restrict__ fB,
                          unsigned short* __restrict__ stage,
                          unsigned short* __restrict__ hn,
                          const float* __restrict__ wtil, int wbase,
                          float& out_acc, int lane, int tid) {
  v8f acc[8] = {};
  const int d   = lane & 15;
  const int hi8 = (lane & 16) ? 8 : 0;
  const unsigned short* xrow = xT + d * HSTRIDE;
  const unsigned short* hrow = hT + d * HSTRIDE;

  uint4* sdst = (uint4*)stage;

  // A-build + 8 WMMAs for one K-step, reading B fragments from the stage
  auto kstep = [&](int ks) {
    union { v16bf v; v2bf h2[8]; } a;
    #pragma unroll
    for (int g = 0; g < 2; ++g) {
      int c = ks * 32 + hi8 + g * 16;      // K base of this group
      int i = c >> 6;
      int j = c & 63;                      // multiple of 8 -> 16B aligned
      union { unsigned u; v2bf v; } xp;
      xp.u = (unsigned)xrow[i] * 0x10001u; // broadcast x[i,d] to both halves
      union { uint4 q; v2bf v[4]; } hv;
      hv.q = *(const uint4*)(hrow + j);    // h[j..j+7, d] packed bf16
      #pragma unroll
      for (int k2 = 0; k2 < 4; ++k2) a.h2[g * 4 + k2] = xp.v * hv.v[k2];
    }
    const v16bf* bp = (const v16bf*)(stage + lane * 16);
    #pragma unroll
    for (int nt = 0; nt < 8; ++nt) {
      v16bf bv = bp[nt * 32];              // halves offset nt*512
      acc[nt] = __builtin_amdgcn_wmma_f32_16x16x32_bf16(
          false, a.v, false, bv, (short)0, acc[nt], false, false);
    }
  };

  // prologue: prefetch K-step 0; steady state prefetches ks+1 unconditionally
  const uint4* n0 = (const uint4*)fB + tid;
  const uint4* n1 = n0 + 256;
  uint4 r0 = *n0;
  uint4 r1 = *n1;
  n0 += FRAG_UINT4_PER_KS;
  n1 += FRAG_UINT4_PER_KS;

  #pragma clang loop unroll(disable)
  for (int ks = 0; ks < KT - 1; ++ks) {
    __syncthreads();                       // all waves done reading old stage
    sdst[tid]       = r0;
    sdst[tid + 256] = r1;
    __syncthreads();
    r0 = *n0;                              // issue next K-step's loads now;
    r1 = *n1;                              // latency hidden by WMMAs below
    n0 += FRAG_UINT4_PER_KS;
    n1 += FRAG_UINT4_PER_KS;
    kstep(ks);
  }
  // peeled last K-step (no prefetch)
  __syncthreads();
  sdst[tid]       = r0;
  sdst[tid + 256] = r1;
  __syncthreads();
  kstep(KT - 1);

  // ---- postprocess: C layout VGPR r, lane L -> (d = r+8*(L>=16), l-col = L&15)
  #pragma unroll
  for (int nt = 0; nt < 8; ++nt) {
    int l = nt * 16 + d;
    bool direct = (!KEEPHALF) || (l >= 64);
    float w = 0.0f;
    if (direct) w = wtil[KEEPHALF ? (wbase + l - 64) : (wbase + l)];
    #pragma unroll
    for (int r = 0; r < 8; ++r) {
      float v = acc[nt][r];
      v = v > 0.0f ? v : 0.0f;             // relu
      if (direct) out_acc += v * w;
      else        hn[(r + hi8) * HSTRIDE + l] = f2bf(v);  // next hidden j = l
    }
  }
}

// ---------------------------------------------------------------------------
// Main kernel: 256 threads = 8 waves, one sample per wave.
// LDS: 8*3*1152 halves (55296B) + 8KB stage + 1KB weights = 64512B.
// ---------------------------------------------------------------------------
__global__ __launch_bounds__(256)
void cin_main(const float* __restrict__ nn_input,
              const float* __restrict__ w_nn,
              const float* __restrict__ b_nn,
              const unsigned short* __restrict__ f0b,
              const unsigned short* __restrict__ f1b,
              const unsigned short* __restrict__ f2b,
              float* __restrict__ out) {
  __shared__ __align__(16) unsigned short sh[8 * 3 * SMP_H];
  __shared__ __align__(16) unsigned short stage[FRAG_HALVES_PER_KS];
  __shared__ float wtil[256];

  const int tid  = threadIdx.x;
  const int wave = tid >> 5;
  const int lane = tid & 31;
  const int s    = blockIdx.x * 8 + wave;

  wtil[tid] = 1.0f + w_nn[tid];            // base_score fold: 1 + w_nn[l]

  unsigned short* xa = sh + wave * 3 * SMP_H;
  unsigned short* hA = xa + SMP_H;
  unsigned short* hB = hA + SMP_H;

  // zero-pad x buffer with wide stores (j in [39,64) must be 0 for layer 0),
  // then transpose-store x0 as bf16 d-major: xa[d*HSTRIDE + i].
  // (DS ops are in-order per wave, so no barrier needed between phases.)
  {
    uint4 z = {0u, 0u, 0u, 0u};
    uint4* xz = (uint4*)xa;                // SMP_H*2/16 = 144 uint4
    for (int t = lane; t < (SMP_H * 2) / 16; t += 32) xz[t] = z;
  }
  const float* xin = nn_input + (size_t)s * XLEN;
  for (int t = lane; t < XLEN; t += 32) {
    int i = t >> 4, d = t & 15;            // x0[b,i,d] at i*16+d
    xa[d * HSTRIDE + i] = f2bf(xin[t]);
  }
  __syncthreads();

  float out_acc = 0.0f;
  run_layer<1>(xa, xa, f0b, stage, hA, wtil, 0,   out_acc, lane, tid);
  run_layer<1>(xa, hA, f1b, stage, hB, wtil, 64,  out_acc, lane, tid);
  run_layer<0>(xa, hB, f2b, stage, hB, wtil, 128, out_acc, lane, tid);

  // wave32 reduction of per-lane partials
  #pragma unroll
  for (int off = 16; off; off >>= 1) out_acc += __shfl_xor(out_acc, off, 32);
  if (lane == 0) out[s] = out_acc + b_nn[0];
}

// ---------------------------------------------------------------------------
extern "C" void kernel_launch(void* const* d_in, const int* in_sizes, int n_in,
                              void* d_out, int out_size, void* d_ws, size_t ws_size,
                              hipStream_t stream) {
  const float* nn_input = (const float*)d_in[0];
  const float* f0   = (const float*)d_in[1];
  const float* f1   = (const float*)d_in[2];
  const float* f2   = (const float*)d_in[3];
  const float* w_nn = (const float*)d_in[4];
  const float* b_nn = (const float*)d_in[5];

  const size_t frag_halves = (size_t)KT * FRAG_HALVES_PER_KS; // 319488
  unsigned short* f0b = (unsigned short*)d_ws;
  unsigned short* f1b = f0b + frag_halves;
  unsigned short* f2b = f1b + frag_halves;
  // workspace: 3 * 319488 * 2 B ~= 1.83 MB

  int n = (int)frag_halves;
  prep_filter<<<(n + 255) / 256, 256, 0, stream>>>(f0, f0b, 0);
  prep_filter<<<(n + 255) / 256, 256, 0, stream>>>(f1, f1b, 1);
  prep_filter<<<(n + 255) / 256, 256, 0, stream>>>(f2, f2b, 1);

  cin_main<<<NB / 8, 256, 0, stream>>>(nn_input, w_nn, b_nn, f0b, f1b, f2b,
                                       (float*)d_out);
}
